// DynamicTimeWarping_4131758539522
// MI455X (gfx1250) — matrix operand order repaired and checked by reading, not compiled, not run on hardware
//
#include <hip/hip_runtime.h>
#include <hip/hip_bf16.h>
#include <math.h>

// ---------------- problem constants ----------------
#define BN   16
#define AN   1024
#define VN   512
#define DD   512
#define NEGV (-1e30f)

// ---------------- WMMA types ----------------
typedef __attribute__((ext_vector_type(16))) __bf16 v16bf;
typedef __attribute__((ext_vector_type(8)))  float  v8f;

union Frag {
    v16bf v;
    unsigned int   u[8];
    unsigned short s[16];
};

__device__ __forceinline__ unsigned short f2bf(float f) {
    union { float f; unsigned int u; } c; c.f = f;
    unsigned int u = c.u + 0x7FFFu + ((c.u >> 16) & 1u);   // RNE
    return (unsigned short)(u >> 16);
}

// =====================================================================
// K0a: fp32 -> bf16 elementwise convert (used for W)
// =====================================================================
__global__ __launch_bounds__(256) void cvt_bf16_kernel(
    const float* __restrict__ in, unsigned short* __restrict__ out, int n)
{
    int i = blockIdx.x * 256 + threadIdx.x;
    if (i < n) out[i] = f2bf(in[i]);
}

// =====================================================================
// K0b: video fp32 [b][v][d] -> bf16 transposed [b][d][v]
// 32x32 LDS tiles; coalesced read and write.
// =====================================================================
__global__ __launch_bounds__(256) void cvt_transpose_kernel(
    const float* __restrict__ video, unsigned short* __restrict__ vidT)
{
    int bid = blockIdx.x;
    int tx = bid & 15; bid >>= 4;   // d tile
    int ty = bid & 15; bid >>= 4;   // v tile
    int b  = bid;

    __shared__ unsigned short T[32][34];   // pad keeps dword alignment, avoids conflicts
    int tid = threadIdx.x;
    int r = tid >> 5, c = tid & 31;

    const float* src = video + ((size_t)b * VN + ty * 32) * DD + tx * 32;
#pragma unroll
    for (int rr = r; rr < 32; rr += 8)
        T[rr][c] = f2bf(src[(size_t)rr * DD + c]);
    __syncthreads();

    unsigned short* dst = vidT + ((size_t)b * DD + tx * 32) * VN + ty * 32;
#pragma unroll
    for (int rr = r; rr < 32; rr += 8)
        dst[(size_t)rr * VN + c] = T[c][rr];
}

// =====================================================================
// K1: projection GEMM (X @ W^T + b) fused with row L2-normalize.
// One block = 16 rows x 512 cols, 8 waves, each wave 4 col-tiles of 16.
// W already bf16 -> inner loop is dword loads + wmma only.
// =====================================================================
__global__ __launch_bounds__(256) void proj_norm_kernel(
    const float* __restrict__ audio, const float* __restrict__ video,
    const unsigned short* __restrict__ Wb, const float* __restrict__ bias,
    unsigned short* __restrict__ a_n, unsigned short* __restrict__ v_n)
{
    const int AUDIO_TILES = (BN * AN) / 16;   // 1024
    int rt = blockIdx.x;
    const float* src;
    unsigned short* dst;
    int rowBase;
    if (rt < AUDIO_TILES) { src = audio; dst = a_n; rowBase = rt * 16; }
    else                  { src = video; dst = v_n; rowBase = (rt - AUDIO_TILES) * 16; }

    __shared__ unsigned short Xs[16][DD + 2];   // bf16 A tile, rows dword-aligned
    __shared__ float rowsq[16];

    int tid = threadIdx.x;
    for (int idx = tid; idx < 16 * DD; idx += 256) {
        int r = idx >> 9, c = idx & (DD - 1);
        Xs[r][c] = f2bf(src[(size_t)(rowBase + r) * DD + c]);
    }
    if (tid < 16) rowsq[tid] = 0.0f;
    __syncthreads();

    int wave = tid >> 5, lane = tid & 31;
    int h = lane >> 4, ln = lane & 15;

    v8f acc[4] = {};
    for (int kb = 0; kb < DD; kb += 32) {
        Frag afr;
#pragma unroll
        for (int g = 0; g < 8; ++g) {
            int k0 = kb + ((g >= 4) ? 16 : 0) + h * 8 + (g & 3) * 2;   // even
            afr.u[g] = *reinterpret_cast<const unsigned int*>(&Xs[ln][k0]);
        }
#pragma unroll
        for (int t = 0; t < 4; ++t) {
            int n = (wave * 4 + t) * 16 + ln;                  // output column e
            const unsigned short* wrow = Wb + (size_t)n * DD;  // W[e][*] bf16
            Frag bfr;
#pragma unroll
            for (int g = 0; g < 8; ++g) {
                int k = kb + ((g >= 4) ? 16 : 0) + h * 8 + (g & 3) * 2;   // even
                bfr.u[g] = *reinterpret_cast<const unsigned int*>(wrow + k);
            }
            acc[t] = __builtin_amdgcn_wmma_f32_16x16x32_bf16(
                false, afr.v, false, bfr.v, (short)0, acc[t], false, false);
        }
    }

    // bias + accumulate sum of squares per row (LDS float atomics)
#pragma unroll
    for (int t = 0; t < 4; ++t) {
        int n = (wave * 4 + t) * 16 + ln;
        float bv = bias[n];
#pragma unroll
        for (int r = 0; r < 8; ++r) {
            float v = acc[t][r] + bv;
            acc[t][r] = v;
            atomicAdd(&rowsq[h * 8 + r], v * v);
        }
    }
    __syncthreads();

    float scale[8];
#pragma unroll
    for (int r = 0; r < 8; ++r)
        scale[r] = 1.0f / fmaxf(sqrtf(rowsq[h * 8 + r]), 1e-12f);

#pragma unroll
    for (int t = 0; t < 4; ++t) {
        int n = (wave * 4 + t) * 16 + ln;
#pragma unroll
        for (int r = 0; r < 8; ++r)
            dst[(size_t)(rowBase + h * 8 + r) * DD + n] = f2bf(acc[t][r] * scale[r]);
    }
}

// =====================================================================
// K2: similarity GEMM: sim[b,a,v] = a_n[b,a,:] . v_n[b,v,:]  (bf16 WMMA)
// grid = 16 * 64 * 4; block = 256 (8 waves x one 16x16 tile each)
// =====================================================================
__global__ __launch_bounds__(256) void sim_kernel(
    const unsigned short* __restrict__ a_n,
    const unsigned short* __restrict__ v_n,
    float* __restrict__ sim)
{
    int bid = blockIdx.x;
    int cblk = bid & 3;  bid >>= 2;
    int mt   = bid & 63; bid >>= 6;
    int b    = bid;

    int tid = threadIdx.x, wave = tid >> 5, lane = tid & 31;
    int h = lane >> 4, ln = lane & 15;

    __shared__ unsigned short As[16][DD + 2];
    const unsigned short* arow = a_n + ((size_t)b * AN + mt * 16) * DD;
    for (int idx = tid; idx < 16 * DD; idx += 256) {
        int r = idx >> 9, c = idx & (DD - 1);
        As[r][c] = arow[(size_t)r * DD + c];
    }
    __syncthreads();

    int n = cblk * 128 + wave * 16 + ln;                 // video index v
    const unsigned short* vrow = v_n + ((size_t)b * VN + n) * DD;

    v8f acc = {};
    for (int kb = 0; kb < DD; kb += 32) {
        Frag afr, bfr;
#pragma unroll
        for (int g = 0; g < 8; ++g) {
            int k0 = kb + ((g >= 4) ? 16 : 0) + h * 8 + (g & 3) * 2;
            afr.u[g] = *reinterpret_cast<const unsigned int*>(&As[ln][k0]);
            int k = kb + ((g >= 4) ? 16 : 0) + h * 8 + (g & 3) * 2;
            bfr.u[g] = *reinterpret_cast<const unsigned int*>(vrow + k);
        }
        acc = __builtin_amdgcn_wmma_f32_16x16x32_bf16(
            false, afr.v, false, bfr.v, (short)0, acc, false, false);
    }

    float* out = sim + ((size_t)b * AN + mt * 16) * VN;
#pragma unroll
    for (int r = 0; r < 8; ++r)
        out[(size_t)(h * 8 + r) * VN + n] = acc[r];
}

// =====================================================================
// K3: DTW via anti-diagonal wavefront. One 512-thread block per batch.
// dp[i,j] = sim[i-1,j-1] + max(dp[i-1,j], dp[i,j-1], dp[i-1,j-1])
// =====================================================================
__global__ __launch_bounds__(512) void dtw_kernel(
    const float* __restrict__ sim, float* __restrict__ score)
{
    int b = blockIdx.x;
    const float* S = sim + (size_t)b * AN * VN;

    __shared__ float buf[3][VN + 1];     // rotating diagonals d%3
    int tid = threadIdx.x;

    for (int j = tid; j <= VN; j += 512) {
        buf[0][j] = (j == 0) ? 0.0f : NEGV;   // diag 0
        buf[1][j] = NEGV;                     // diag 1
    }
    __syncthreads();

    int j = tid + 1;                          // this thread's column, 1..512
    for (int d = 2; d <= AN + VN; ++d) {
        float* C  = buf[d % 3];
        const float* P1 = buf[(d - 1) % 3];
        const float* P2 = buf[(d - 2) % 3];
        int lo = (d - AN > 1) ? (d - AN) : 1;
        int hi = (d - 1 < VN) ? (d - 1) : VN;
        if (j >= lo && j <= hi) {
            int i = d - j;
            float up   = P1[j];        // (i-1, j)
            float left = P1[j - 1];    // (i, j-1)
            float diag = P2[j - 1];    // (i-1, j-1)
            C[j] = S[(size_t)(i - 1) * VN + (j - 1)] + fmaxf(fmaxf(up, left), diag);
        } else {
            C[j] = NEGV;
        }
        if (tid == 0) C[0] = NEGV;
        __syncthreads();
    }
    if (tid == 0) score[b] = buf[(AN + VN) % 3][VN];
}

// =====================================================================
// K4: row softmax over V, write bf16 prob. One wave per row.
// =====================================================================
__global__ __launch_bounds__(256) void softmax_kernel(
    const float* __restrict__ sim, unsigned short* __restrict__ prob)
{
    int row  = blockIdx.x * 8 + (threadIdx.x >> 5);   // 16384 rows total
    int lane = threadIdx.x & 31;
    const float* s = sim + (size_t)row * VN;

    float x[16];
    float mx = NEGV;
#pragma unroll
    for (int i = 0; i < 16; ++i) {
        x[i] = s[lane + i * 32];
        mx = fmaxf(mx, x[i]);
    }
#pragma unroll
    for (int off = 16; off > 0; off >>= 1)
        mx = fmaxf(mx, __shfl_xor(mx, off, 32));

    float sum = 0.0f;
#pragma unroll
    for (int i = 0; i < 16; ++i) {
        x[i] = __expf(x[i] - mx);
        sum += x[i];
    }
#pragma unroll
    for (int off = 16; off > 0; off >>= 1)
        sum += __shfl_xor(sum, off, 32);

    float inv = 1.0f / sum;
    unsigned short* p = prob + (size_t)row * VN;
#pragma unroll
    for (int i = 0; i < 16; ++i)
        p[lane + i * 32] = f2bf(x[i] * inv);
}

// =====================================================================
// K5: aligned_video[b,a,d] = sum_v prob[b,a,v] * video[b,v,d] (bf16 WMMA)
// B matrix read from transposed vidT[b][d][v] -> contiguous dword pairs.
// =====================================================================
__global__ __launch_bounds__(256) void aligned_kernel(
    const unsigned short* __restrict__ prob,
    const unsigned short* __restrict__ vidT,
    float* __restrict__ outp)
{
    int bid = blockIdx.x;
    int cblk = bid & 3;  bid >>= 2;
    int mt   = bid & 63; bid >>= 6;
    int b    = bid;

    int tid = threadIdx.x, wave = tid >> 5, lane = tid & 31;
    int h = lane >> 4, ln = lane & 15;

    __shared__ unsigned short As[16][VN + 2];
    const unsigned short* prow = prob + ((size_t)b * AN + mt * 16) * VN;
    for (int idx = tid; idx < 16 * VN; idx += 256) {
        int r = idx >> 9, c = idx & (VN - 1);
        As[r][c] = prow[(size_t)r * VN + c];
    }
    __syncthreads();

    int n = cblk * 128 + wave * 16 + ln;                 // feature dim d
    const unsigned short* vrow = vidT + ((size_t)b * DD + n) * VN;  // [d][v]

    v8f acc = {};
    for (int kb = 0; kb < VN; kb += 32) {
        Frag afr, bfr;
#pragma unroll
        for (int g = 0; g < 8; ++g) {
            int k0 = kb + ((g >= 4) ? 16 : 0) + h * 8 + (g & 3) * 2;
            afr.u[g] = *reinterpret_cast<const unsigned int*>(&As[ln][k0]);
            int k = kb + ((g >= 4) ? 16 : 0) + h * 8 + (g & 3) * 2;   // video row v (even)
            bfr.u[g] = *reinterpret_cast<const unsigned int*>(vrow + k);
        }
        acc = __builtin_amdgcn_wmma_f32_16x16x32_bf16(
            false, afr.v, false, bfr.v, (short)0, acc, false, false);
    }

    float* out = outp + ((size_t)b * AN + mt * 16) * DD;
#pragma unroll
    for (int r = 0; r < 8; ++r)
        out[(size_t)(h * 8 + r) * DD + n] = acc[r];
}

// =====================================================================
// launch
// =====================================================================
extern "C" void kernel_launch(void* const* d_in, const int* in_sizes, int n_in,
                              void* d_out, int out_size, void* d_ws, size_t ws_size,
                              hipStream_t stream) {
    const float* audio = (const float*)d_in[0];   // [16,1024,512]
    const float* video = (const float*)d_in[1];   // [16,512,512]
    const float* W     = (const float*)d_in[2];   // [512,512]
    const float* bias  = (const float*)d_in[3];   // [512]
    float* out = (float*)d_out;                   // aligned (8388608) + scores (16)

    char* ws = (char*)d_ws;
    unsigned short* v_n  = (unsigned short*)(ws + 0);          //  8.4 MB
    unsigned short* vidT = (unsigned short*)(ws + 8388608);    //  8.4 MB (transposed)
    float*          sim  = (float*)        (ws + 16777216);    // 33.6 MB
    unsigned short* a_n  = (unsigned short*)(ws + 50331648);   // 16.8 MB
    unsigned short* Wb   = (unsigned short*)(ws + 67108864);   //  0.5 MB
    unsigned short* prob = a_n;   // a_n dead after sim_kernel; reuse within this call

    // K0a: W -> bf16 (once; L2-resident for all projection blocks)
    cvt_bf16_kernel<<<(DD * DD) / 256, 256, 0, stream>>>(W, Wb, DD * DD);
    // K0b: video -> bf16 transposed [b][d][v]
    cvt_transpose_kernel<<<BN * 16 * 16, 256, 0, stream>>>(video, vidT);
    // K1: projections + normalize (audio tiles then video tiles)
    proj_norm_kernel<<<(BN * AN) / 16 + (BN * VN) / 16, 256, 0, stream>>>(
        audio, video, Wb, bias, a_n, v_n);
    // K2: similarity
    sim_kernel<<<BN * (AN / 16) * 4, 256, 0, stream>>>(a_n, v_n, sim);
    // K3: DTW scores -> tail of d_out
    dtw_kernel<<<BN, 512, 0, stream>>>(sim, out + (size_t)BN * AN * DD);
    // K4: softmax -> bf16 prob
    softmax_kernel<<<(BN * AN) / 8, 256, 0, stream>>>(sim, prob);
    // K5: aligned video -> d_out
    aligned_kernel<<<BN * (AN / 16) * 4, 256, 0, stream>>>(prob, vidT, out);
}
